// VDPWIModel_4037269259126
// MI455X (gfx1250) — matrix operand order repaired and checked by reading, not compiled, not run on hardware
//
#include <hip/hip_runtime.h>
#include <hip/hip_bf16.h>
#include <cstdint>

typedef __attribute__((ext_vector_type(16))) _Float16 v16h;
typedef __attribute__((ext_vector_type(8)))  _Float16 v8h;
typedef __attribute__((ext_vector_type(8)))  float    v8f;

#define NEGC (-10000.0f)

// ---------------------------------------------------------------- utilities

__global__ void zero_u32(uint32_t* p, size_t n) {
  size_t i  = (size_t)blockIdx.x * blockDim.x + threadIdx.x;
  size_t st = (size_t)gridDim.x * blockDim.x;
  for (; i < n; i += st) p[i] = 0u;
}

// sent [B,D,L] f32 -> [B*L, 320] f16 (K padded 300->320, pad pre-zeroed)
__global__ void k_cvt_sent(const float* __restrict__ s, _Float16* __restrict__ dst) {
  int idx = blockIdx.x * blockDim.x + threadIdx.x;
  if (idx >= 256 * 32 * 300) return;
  int d = idx % 300; int t2 = idx / 300; int t = t2 & 31; int b = t2 >> 5;
  dst[((size_t)(b * 32 + t)) * 320 + d] = (_Float16)s[((size_t)b * 300 + d) * 32 + t];
}

// generic [N,K] f32 -> f16 with row stride ldd (pads pre-zeroed)
__global__ void k_cvt_mat(const float* __restrict__ src, _Float16* __restrict__ dst,
                          int N, int K, int ldd) {
  int idx = blockIdx.x * blockDim.x + threadIdx.x;
  if (idx >= N * K) return;
  int n = idx / K; int k = idx - n * K;
  dst[(size_t)n * ldd + k] = (_Float16)src[idx];
}

__global__ void k_bsum(const float* a, const float* b, float* o) {
  int i = blockIdx.x * blockDim.x + threadIdx.x;
  if (i < 1000) o[i] = a[i] + b[i];
}

__global__ void k_copyf(const float* s, float* d, int n) {
  int i = blockIdx.x * blockDim.x + threadIdx.x;
  if (i < n) d[i] = s[i];
}

// OIHW conv weights -> [tap][Npad][Cpad] f16 (pads pre-zeroed)
__global__ void k_cvt_convw(const float* __restrict__ src, _Float16* __restrict__ dst,
                            int Cout, int Cin, int Npad, int Cpad) {
  int idx = blockIdx.x * blockDim.x + threadIdx.x;
  int total = 9 * Cout * Cin;
  if (idx >= total) return;
  int c = idx % Cin; int t2 = idx / Cin; int n = t2 % Cout; int tap = t2 / Cout;
  int dr = tap / 3, ds = tap - dr * 3;
  dst[((size_t)tap * Npad + n) * Cpad + c] =
      (_Float16)src[(((size_t)n * Cin + c) * 3 + dr) * 3 + ds];
}

// ------------------------------------------------------------- WMMA helpers
// A fragment (16x32 f16): lane<16 holds row r, K {0..15}; lane>=16 K {16..31}
// B fragment (32x16 f16): lane holds column n=lane%16, contiguous K chunk of 16

__device__ __forceinline__ v16h load_a(const _Float16* rowBase, int hi) {
  union { v16h v; v8h h[2]; } u;
  u.h[0] = *(const v8h*)(rowBase + hi * 16);
  u.h[1] = *(const v8h*)(rowBase + hi * 16 + 8);
  return u.v;
}

// ---------------------------------------------------------------- LSTM GEMMs

// Xpre = s_h[8192x320] @ W[1024x320]^T -> f32 [8192][1024]
__global__ void k_xpre(const _Float16* __restrict__ A, const _Float16* __restrict__ Bw,
                       float* __restrict__ out) {
  int wave = blockIdx.x * (blockDim.x >> 5) + (threadIdx.x >> 5);
  int nT = wave & 63; int mT = wave >> 6;
  int lane = threadIdx.x & 31; int r = lane & 15; int hi = (lane >> 4) & 1;
  const _Float16* arow = A  + ((size_t)(mT * 16 + r)) * 320;
  const _Float16* brow = Bw + ((size_t)(nT * 16 + r)) * 320 + hi * 16;
  v8f acc = {};
#pragma unroll
  for (int k0 = 0; k0 < 320; k0 += 32) {
    v16h a = load_a(arow + k0, hi);
    v16h b = *(const v16h*)(brow + k0);
    acc = __builtin_amdgcn_wmma_f32_16x16x32_f16(false, a, false, b, (short)0, acc, false, false);
  }
  int n = nT * 16 + r;
  size_t mB = (size_t)mT * 16 + hi * 8;
#pragma unroll
  for (int v = 0; v < 8; ++v) out[(mB + v) * 1024 + n] = acc[v];
}

// G = h[256x256] @ Whh[1024x256]^T + Xpre[b*32+tsrc] + bsum
__global__ void k_rec(const _Float16* __restrict__ hb, const _Float16* __restrict__ whh,
                      const float* __restrict__ xpre, const float* __restrict__ bsum,
                      float* __restrict__ G, int tsrc) {
  int wave = blockIdx.x * (blockDim.x >> 5) + (threadIdx.x >> 5);
  int nT = wave & 63; int mT = wave >> 6;
  int lane = threadIdx.x & 31; int r = lane & 15; int hi = (lane >> 4) & 1;
  const _Float16* arow = hb  + ((size_t)(mT * 16 + r)) * 256;
  const _Float16* brow = whh + ((size_t)(nT * 16 + r)) * 256 + hi * 16;
  v8f acc = {};
#pragma unroll
  for (int k0 = 0; k0 < 256; k0 += 32) {
    v16h a = load_a(arow + k0, hi);
    v16h b = *(const v16h*)(brow + k0);
    acc = __builtin_amdgcn_wmma_f32_16x16x32_f16(false, a, false, b, (short)0, acc, false, false);
  }
  int n = nT * 16 + r; float bs = bsum[n];
#pragma unroll
  for (int v = 0; v < 8; ++v) {
    int b = mT * 16 + v + hi * 8;
    G[(size_t)b * 1024 + n] = acc[v] + xpre[((size_t)b * 32 + tsrc) * 1024 + n] + bs;
  }
}

// gate order i,f,g,o at offsets 0,250,500,750
__global__ void k_gates(const float* __restrict__ G, float* __restrict__ cst,
                        _Float16* __restrict__ hb, float* __restrict__ hs, int t) {
  int idx = blockIdx.x * blockDim.x + threadIdx.x;
  if (idx >= 256 * 250) return;
  int b = idx / 250; int h = idx - b * 250;
  const float* g = G + (size_t)b * 1024;
  float gi = g[h], gf = g[250 + h], gg = g[500 + h], go = g[750 + h];
  float si = 1.f / (1.f + expf(-gi));
  float sf = 1.f / (1.f + expf(-gf));
  float so = 1.f / (1.f + expf(-go));
  float c = sf * cst[idx] + si * tanhf(gg);
  float hn = so * tanhf(c);
  cst[idx] = c;
  hb[(size_t)b * 256 + h] = (_Float16)hn;
  hs[((size_t)b * 32 + t) * 250 + h] = hn;
}

// -------------------------------------------------------------- similarities

__global__ void k_norms(const float* __restrict__ hF, const float* __restrict__ hB,
                        float* __restrict__ sF, float* __restrict__ sB, float* __restrict__ sS) {
  int idx = blockIdx.x * blockDim.x + threadIdx.x;
  if (idx >= 8192) return;
  const float* f = hF + (size_t)idx * 250;
  const float* b = hB + (size_t)idx * 250;
  float a0 = 0.f, a1 = 0.f, a2 = 0.f;
  for (int k = 0; k < 250; ++k) {
    float fv = f[k], bv = b[k], sv = fv + bv;
    a0 += fv * fv; a1 += bv * bv; a2 += sv * sv;
  }
  sF[idx] = a0; sB[idx] = a1; sS[idx] = a2;
}

__global__ void k_sims(const float* __restrict__ hF1, const float* __restrict__ hB1,
                       const float* __restrict__ hF2, const float* __restrict__ hB2,
                       const float* __restrict__ sF1, const float* __restrict__ sB1,
                       const float* __restrict__ sS1, const float* __restrict__ sF2,
                       const float* __restrict__ sB2, const float* __restrict__ sS2,
                       const int* __restrict__ len1, const int* __restrict__ len2,
                       float* __restrict__ simm) {
  int idx = blockIdx.x * blockDim.x + threadIdx.x;
  if (idx >= 256 * 1024) return;
  int b = idx >> 10; int ij = idx & 1023; int i = ij >> 5; int j = ij & 31;
  const float* f1 = hF1 + ((size_t)b * 32 + i) * 250;
  const float* b1 = hB1 + ((size_t)b * 32 + i) * 250;
  const float* f2 = hF2 + ((size_t)b * 32 + j) * 250;
  const float* b2 = hB2 + ((size_t)b * 32 + j) * 250;
  float dFF = 0.f, dBB = 0.f, dFB = 0.f, dBF = 0.f;
  for (int k = 0; k < 250; ++k) {
    float a = f1[k], c = b1[k], d = f2[k], e = b2[k];
    dFF += a * d; dBB += c * e; dFB += a * e; dBF += c * d;
  }
  float vF1 = sF1[b * 32 + i], vB1 = sB1[b * 32 + i], vS1 = sS1[b * 32 + i];
  float vF2 = sF2[b * 32 + j], vB2 = sB2[b * 32 + j], vS2 = sS2[b * 32 + j];
  float padv = ((i >= len1[b]) || (j >= len2[b])) ? 1.f : 0.f;
  float* base = simm + ((size_t)b * 12) * 1024 + ij;
#define PUT(ch, val) base[(size_t)(ch) * 1024] = (val) + NEGC * padv
  float dcat = dFF + dBB;
  float nc1 = sqrtf(vF1 + vB1), nc2 = sqrtf(vF2 + vB2);
  PUT(0, dcat);
  PUT(1, dcat / (nc1 * nc2 + 1e-8f));
  PUT(2, sqrtf(fmaxf(vF1 + vB1 + vF2 + vB2 - 2.f * dcat, 1e-12f)));
  float nf1 = sqrtf(vF1), nf2 = sqrtf(vF2);
  PUT(3, dFF);
  PUT(4, dFF / (nf1 * nf2 + 1e-8f));
  PUT(5, sqrtf(fmaxf(vF1 + vF2 - 2.f * dFF, 1e-12f)));
  float nb1 = sqrtf(vB1), nb2 = sqrtf(vB2);
  PUT(6, dBB);
  PUT(7, dBB / (nb1 * nb2 + 1e-8f));
  PUT(8, sqrtf(fmaxf(vB1 + vB2 - 2.f * dBB, 1e-12f)));
  float dS = dFF + dBB + dFB + dBF;
  float ns1 = sqrtf(vS1), ns2 = sqrtf(vS2);
  PUT(9, dS);
  PUT(10, dS / (ns1 * ns2 + 1e-8f));
  PUT(11, sqrtf(fmaxf(vS1 + vS2 - 2.f * dS, 1e-12f)));
#undef PUT
}

// sequential greedy argmax masking; one 32-lane block per (batch, channel 9/10)
__global__ void k_greedy(const float* __restrict__ simm, unsigned char* __restrict__ sel) {
  int b = blockIdx.x & 255; int which = blockIdx.x >> 8; int ch = 9 + which;
  __shared__ float M[32][33];
  __shared__ float rbest[32];
  __shared__ int   rcol[32];
  __shared__ int   g_r, g_c, g_act;
  int lane = threadIdx.x;
  const float* src = simm + (((size_t)b * 12 + ch) << 10);
  for (int j = 0; j < 32; ++j) M[lane][j] = src[lane * 32 + j];
  unsigned char* selb = sel + (size_t)which * 256 * 1024 + (size_t)b * 1024;
  __syncthreads();
  for (int it = 0; it < 32; ++it) {
    float best = -3.4e38f; int bc = 0;
    for (int j = 0; j < 32; ++j) { float v = M[lane][j]; if (v > best) { best = v; bc = j; } }
    rbest[lane] = best; rcol[lane] = bc;
    __syncthreads();
    if (lane == 0) {
      float gb = -3.4e38f; int gr = 0;
      for (int r2 = 0; r2 < 32; ++r2) { if (rbest[r2] > gb) { gb = rbest[r2]; gr = r2; } }
      g_r = gr; g_c = rcol[gr]; g_act = (gb >= NEGC * 0.5f) ? 1 : 0;
      if (g_act) selb[gr * 32 + rcol[gr]] = 1;
    }
    __syncthreads();
    if (g_act) { M[lane][g_c] = NEGC; M[g_r][lane] = NEGC; }
    __syncthreads();
  }
}

// focus -> channels-last spatially padded f16 input [B][34][34][32]
__global__ void k_focus(const float* __restrict__ simm, const unsigned char* __restrict__ sel9,
                        const unsigned char* __restrict__ sel10, const int* __restrict__ len1,
                        const int* __restrict__ len2, _Float16* __restrict__ in1) {
  int idx = blockIdx.x * blockDim.x + threadIdx.x;
  if (idx >= 256 * 12 * 1024) return;
  int b = idx / (12 * 1024); int rem = idx - b * 12 * 1024;
  int ch = rem >> 10; int ij = rem & 1023; int i = ij >> 5; int j = ij & 31;
  float padv = ((i >= len1[b]) || (j >= len2[b])) ? 1.f : 0.f;
  int s = sel9[(size_t)b * 1024 + ij] | sel10[(size_t)b * 1024 + ij];
  float m = s ? 1.f : 0.1f;
  float v = m * simm[((size_t)b * 12 + ch) * 1024 + ij] * (1.f - padv);
  in1[(((size_t)b * 34 + i + 1) * 34 + (j + 1)) * 32 + ch] = (_Float16)v;
}

// ------------------------------------------------------------- direct conv 3x3
// Templated so the tap x channel-chunk K-loop fully unrolls into a WMMA stream.
// in: [B][Hout+2][Hout+2][Cpad] f16 (zero borders); wt: [tap][Npad][Cpad] f16
// out: [B][Hout][Hout][Npad] f16 = relu(conv + bias)
template <int Hout, int Cpad, int Npad>
__global__ void k_conv(const _Float16* __restrict__ in, const _Float16* __restrict__ wt,
                       const float* __restrict__ bias, _Float16* __restrict__ out) {
  constexpr int HW = Hout * Hout;
  constexpr int Hp = Hout + 2;
  int wave = blockIdx.x * (blockDim.x >> 5) + (threadIdx.x >> 5);
  constexpr int nTiles = Npad >> 4;
  int nT = wave % nTiles; int mT = wave / nTiles;
  int lane = threadIdx.x & 31; int r = lane & 15; int hi = (lane >> 4) & 1;
  int p = mT * 16 + r;
  int bimg = p / HW; int rem = p - bimg * HW; int y = rem / Hout; int x = rem - y * Hout;
  const _Float16* ibase = in + (((size_t)bimg * Hp + y) * Hp + x) * Cpad;
  const _Float16* wlane = wt + ((size_t)(nT * 16 + r)) * Cpad + hi * 16;
  v8f acc = {};
#pragma unroll
  for (int tap = 0; tap < 9; ++tap) {
    const int dr = tap / 3, ds = tap - dr * 3;
    const _Float16* abase = ibase + ((size_t)dr * Hp + ds) * Cpad;
    const _Float16* wbase = wlane + (size_t)tap * Npad * Cpad;
#pragma unroll
    for (int c0 = 0; c0 < Cpad; c0 += 32) {
      v16h a = load_a(abase + c0, hi);
      v16h bfr = *(const v16h*)(wbase + c0);
      acc = __builtin_amdgcn_wmma_f32_16x16x32_f16(false, a, false, bfr, (short)0, acc, false, false);
    }
  }
  int n = nT * 16 + r;
  float bv = bias[n];
#pragma unroll
  for (int v = 0; v < 8; ++v) {
    int m = mT * 16 + v + hi * 8;
    float val = fmaxf(acc[v] + bv, 0.f);
    out[(size_t)m * Npad + n] = (_Float16)val;  // m == b*HW + pixel, row-major
  }
}

// 2x2 maxpool, channels-last; writes into padded dst interior at offset `off`
__global__ void k_pool(const _Float16* __restrict__ in, _Float16* __restrict__ out,
                       int C, int Hin, int Hp, int off) {
  int Hout = Hin >> 1;
  int total = 256 * Hout * Hout * C;
  int idx = blockIdx.x * blockDim.x + threadIdx.x;
  if (idx >= total) return;
  int c = idx % C; int t = idx / C; int x = t % Hout; t /= Hout; int y = t % Hout;
  int b = t / Hout;
  const _Float16* p = in + (((size_t)b * Hin + 2 * y) * Hin + 2 * x) * C + c;
  float v00 = (float)p[0], v01 = (float)p[C];
  float v10 = (float)p[(size_t)Hin * C], v11 = (float)p[(size_t)Hin * C + C];
  float v = fmaxf(fmaxf(v00, v01), fmaxf(v10, v11));
  out[(((size_t)b * Hp + y + off) * Hp + (x + off)) * C + c] = (_Float16)v;
}

// ---------------------------------------------------------------------- FCs

__global__ void k_fc1(const _Float16* __restrict__ A, const _Float16* __restrict__ Bw,
                      const float* __restrict__ bias, _Float16* __restrict__ out) {
  // M=256, K=128, N=128
  int wave = blockIdx.x * (blockDim.x >> 5) + (threadIdx.x >> 5);
  int nT = wave & 7; int mT = wave >> 3;
  int lane = threadIdx.x & 31; int r = lane & 15; int hi = (lane >> 4) & 1;
  const _Float16* arow = A  + ((size_t)(mT * 16 + r)) * 128;
  const _Float16* brow = Bw + ((size_t)(nT * 16 + r)) * 128 + hi * 16;
  v8f acc = {};
#pragma unroll
  for (int k0 = 0; k0 < 128; k0 += 32) {
    v16h a = load_a(arow + k0, hi);
    v16h b = *(const v16h*)(brow + k0);
    acc = __builtin_amdgcn_wmma_f32_16x16x32_f16(false, a, false, b, (short)0, acc, false, false);
  }
  int n = nT * 16 + r; float bv = bias[n];
#pragma unroll
  for (int v = 0; v < 8; ++v) {
    int m = mT * 16 + v + hi * 8;
    out[(size_t)m * 128 + n] = (_Float16)fmaxf(acc[v] + bv, 0.f);
  }
}

__global__ void k_fc2(const _Float16* __restrict__ A, const _Float16* __restrict__ Bw,
                      const float* __restrict__ bias, float* __restrict__ out) {
  // M=256, K=128, N=16 (padded from 5)
  int wave = blockIdx.x * (blockDim.x >> 5) + (threadIdx.x >> 5);
  int mT = wave;
  int lane = threadIdx.x & 31; int r = lane & 15; int hi = (lane >> 4) & 1;
  const _Float16* arow = A  + ((size_t)(mT * 16 + r)) * 128;
  const _Float16* brow = Bw + ((size_t)r) * 128 + hi * 16;
  v8f acc = {};
#pragma unroll
  for (int k0 = 0; k0 < 128; k0 += 32) {
    v16h a = load_a(arow + k0, hi);
    v16h b = *(const v16h*)(brow + k0);
    acc = __builtin_amdgcn_wmma_f32_16x16x32_f16(false, a, false, b, (short)0, acc, false, false);
  }
  int n = r; float bv = bias[n];
#pragma unroll
  for (int v = 0; v < 8; ++v) {
    int m = mT * 16 + v + hi * 8;
    out[(size_t)m * 16 + n] = acc[v] + bv;
  }
}

__global__ void k_lsm(const float* __restrict__ logits, float* __restrict__ out) {
  int b = blockIdx.x * blockDim.x + threadIdx.x;
  if (b >= 256) return;
  float x[5]; float mx = -3.4e38f;
  for (int j = 0; j < 5; ++j) { x[j] = logits[(size_t)b * 16 + j]; mx = fmaxf(mx, x[j]); }
  float s = 0.f;
  for (int j = 0; j < 5; ++j) s += expf(x[j] - mx);
  float l = logf(s);
  for (int j = 0; j < 5; ++j) out[(size_t)b * 5 + j] = x[j] - mx - l;
}

// ------------------------------------------------------------------ launcher

extern "C" void kernel_launch(void* const* d_in, const int* in_sizes, int n_in,
                              void* d_out, int out_size, void* d_ws, size_t ws_size,
                              hipStream_t stream) {
  (void)in_sizes; (void)n_in; (void)out_size; (void)ws_size;
  const float* sent1 = (const float*)d_in[0];
  const float* sent2 = (const float*)d_in[1];
  const int*   len1  = (const int*)d_in[2];
  const int*   len2  = (const int*)d_in[3];
  const float* Wih   = (const float*)d_in[4];
  const float* Whh   = (const float*)d_in[5];
  const float* bih   = (const float*)d_in[6];
  const float* bhh   = (const float*)d_in[7];
  const float* c1w = (const float*)d_in[8],  *c1b = (const float*)d_in[9];
  const float* c2w = (const float*)d_in[10], *c2b = (const float*)d_in[11];
  const float* c3w = (const float*)d_in[12], *c3b = (const float*)d_in[13];
  const float* c4w = (const float*)d_in[14], *c4b = (const float*)d_in[15];
  const float* c5w = (const float*)d_in[16], *c5b = (const float*)d_in[17];
  const float* dnnw = (const float*)d_in[18], *dnnb = (const float*)d_in[19];
  const float* outw = (const float*)d_in[20], *outb = (const float*)d_in[21];

  char* base = (char*)d_ws;
  size_t off = 0;
  auto A = [&](size_t bytes) -> char* {
    char* p = base + off;
    off = (off + bytes + 255) & ~(size_t)255;
    return p;
  };
  _Float16* s1h  = (_Float16*)A(8192ull * 320 * 2);
  _Float16* s2h  = (_Float16*)A(8192ull * 320 * 2);
  _Float16* wihh = (_Float16*)A(1024ull * 320 * 2);
  _Float16* whhh = (_Float16*)A(1024ull * 256 * 2);
  float*    bsum = (float*)A(1024ull * 4);
  float*    xpre1 = (float*)A(8192ull * 1024 * 4);
  float*    xpre2 = (float*)A(8192ull * 1024 * 4);
  float*    hsF1 = (float*)A(256ull * 32 * 250 * 4);
  float*    hsF2 = (float*)A(256ull * 32 * 250 * 4);
  float*    hsB1 = (float*)A(256ull * 32 * 250 * 4);
  float*    hsB2 = (float*)A(256ull * 32 * 250 * 4);
  _Float16* hb[4]; float* cb[4];
  for (int i = 0; i < 4; ++i) hb[i] = (_Float16*)A(256ull * 256 * 2);
  for (int i = 0; i < 4; ++i) cb[i] = (float*)A(256ull * 250 * 4);
  float*    G = (float*)A(256ull * 1024 * 4);
  float *sF1 = (float*)A(8192 * 4), *sB1 = (float*)A(8192 * 4), *sS1 = (float*)A(8192 * 4);
  float *sF2 = (float*)A(8192 * 4), *sB2 = (float*)A(8192 * 4), *sS2 = (float*)A(8192 * 4);
  float*    simm = (float*)A(256ull * 12 * 1024 * 4);
  unsigned char* sel = (unsigned char*)A(2ull * 256 * 1024);
  _Float16* in1  = (_Float16*)A(256ull * 34 * 34 * 32 * 2);
  _Float16* c1wh = (_Float16*)A(9ull * 128 * 32 * 2);
  _Float16* c2wh = (_Float16*)A(9ull * 192 * 128 * 2);
  _Float16* c3wh = (_Float16*)A(9ull * 192 * 192 * 2);
  _Float16* c4wh = (_Float16*)A(9ull * 192 * 192 * 2);
  _Float16* c5wh = (_Float16*)A(9ull * 128 * 192 * 2);
  float*    c2bp = (float*)A(192 * 4);
  float*    outbp = (float*)A(16 * 4);
  _Float16* dnnwh = (_Float16*)A(128ull * 128 * 2);
  _Float16* outwh = (_Float16*)A(16ull * 128 * 2);
  _Float16* out1 = (_Float16*)A(256ull * 32 * 32 * 128 * 2);
  _Float16* in2  = (_Float16*)A(256ull * 18 * 18 * 128 * 2);
  _Float16* out2 = (_Float16*)A(256ull * 16 * 16 * 192 * 2);
  _Float16* in3  = (_Float16*)A(256ull * 10 * 10 * 192 * 2);
  _Float16* out3 = (_Float16*)A(256ull * 8 * 8 * 192 * 2);
  _Float16* in4  = (_Float16*)A(256ull * 6 * 6 * 192 * 2);
  _Float16* out4 = (_Float16*)A(256ull * 4 * 4 * 192 * 2);
  _Float16* in5  = (_Float16*)A(256ull * 4 * 4 * 192 * 2);
  _Float16* out5 = (_Float16*)A(256ull * 2 * 2 * 128 * 2);
  _Float16* pooled = (_Float16*)A(256ull * 128 * 2);
  _Float16* fc1 = (_Float16*)A(256ull * 128 * 2);
  float*    logits = (float*)A(256ull * 16 * 4);

  // 1) zero all used workspace (initializes every pad / h0 / c0 / sel / borders)
  zero_u32<<<8192, 256, 0, stream>>>((uint32_t*)d_ws, off / 4);

  // 2) conversions
  k_cvt_sent<<<9600, 256, 0, stream>>>(sent1, s1h);
  k_cvt_sent<<<9600, 256, 0, stream>>>(sent2, s2h);
  k_cvt_mat<<<(1000 * 300 + 255) / 256, 256, 0, stream>>>(Wih, wihh, 1000, 300, 320);
  k_cvt_mat<<<(1000 * 250 + 255) / 256, 256, 0, stream>>>(Whh, whhh, 1000, 250, 256);
  k_bsum<<<4, 256, 0, stream>>>(bih, bhh, bsum);
  k_cvt_convw<<<(9 * 128 * 12 + 255) / 256, 256, 0, stream>>>(c1w, c1wh, 128, 12, 128, 32);
  k_cvt_convw<<<(9 * 164 * 128 + 255) / 256, 256, 0, stream>>>(c2w, c2wh, 164, 128, 192, 128);
  k_cvt_convw<<<(9 * 192 * 164 + 255) / 256, 256, 0, stream>>>(c3w, c3wh, 192, 164, 192, 192);
  k_cvt_convw<<<(9 * 192 * 192 + 255) / 256, 256, 0, stream>>>(c4w, c4wh, 192, 192, 192, 192);
  k_cvt_convw<<<(9 * 128 * 192 + 255) / 256, 256, 0, stream>>>(c5w, c5wh, 128, 192, 128, 192);
  k_copyf<<<1, 256, 0, stream>>>(c2b, c2bp, 164);
  k_copyf<<<1, 256, 0, stream>>>(outb, outbp, 5);
  k_cvt_mat<<<(128 * 128 + 255) / 256, 256, 0, stream>>>(dnnw, dnnwh, 128, 128, 128);
  k_cvt_mat<<<(5 * 128 + 255) / 256, 256, 0, stream>>>(outw, outwh, 5, 128, 128);

  // 3) input projections (time-parallel WMMA GEMMs)
  k_xpre<<<8192, 128, 0, stream>>>(s1h, wihh, xpre1);
  k_xpre<<<8192, 128, 0, stream>>>(s2h, wihh, xpre2);

  // 4) 4 LSTM passes, 32 sequential steps each
  struct P { float* xp; int fwd; float* hs; _Float16* h; float* c; };
  P passes[4] = { { xpre1, 1, hsF1, hb[0], cb[0] },
                  { xpre2, 1, hsF2, hb[1], cb[1] },
                  { xpre1, 0, hsB1, hb[2], cb[2] },
                  { xpre2, 0, hsB2, hb[3], cb[3] } };
  for (int p = 0; p < 4; ++p) {
    for (int t = 0; t < 32; ++t) {
      int tsrc = passes[p].fwd ? t : 31 - t;
      k_rec<<<256, 128, 0, stream>>>(passes[p].h, whhh, passes[p].xp, bsum, G, tsrc);
      k_gates<<<250, 256, 0, stream>>>(G, passes[p].c, passes[p].h, passes[p].hs, t);
    }
  }

  // 5) similarities + greedy focus
  k_norms<<<32, 256, 0, stream>>>(hsF1, hsB1, sF1, sB1, sS1);
  k_norms<<<32, 256, 0, stream>>>(hsF2, hsB2, sF2, sB2, sS2);
  k_sims<<<1024, 256, 0, stream>>>(hsF1, hsB1, hsF2, hsB2, sF1, sB1, sS1, sF2, sB2, sS2,
                                   len1, len2, simm);
  k_greedy<<<512, 32, 0, stream>>>(simm, sel);
  k_focus<<<12288, 256, 0, stream>>>(simm, sel, sel + 256 * 1024, len1, len2, in1);

  // 6) conv stack (direct WMMA conv + fused bias/relu, maxpool into padded next input)
  k_conv<32, 32, 128><<<32768, 128, 0, stream>>>(in1, c1wh, c1b, out1);
  k_pool<<<32768, 256, 0, stream>>>(out1, in2, 128, 32, 18, 1);
  k_conv<16, 128, 192><<<12288, 128, 0, stream>>>(in2, c2wh, c2bp, out2);
  k_pool<<<12288, 256, 0, stream>>>(out2, in3, 192, 16, 10, 1);
  k_conv<8, 192, 192><<<3072, 128, 0, stream>>>(in3, c3wh, c3b, out3);
  k_pool<<<3072, 256, 0, stream>>>(out3, in4, 192, 8, 6, 1);
  k_conv<4, 192, 192><<<768, 128, 0, stream>>>(in4, c4wh, c4b, out4);
  k_pool<<<768, 256, 0, stream>>>(out4, in5, 192, 4, 4, 1);
  k_conv<2, 192, 128><<<128, 128, 0, stream>>>(in5, c5wh, c5b, out5);
  k_pool<<<128, 256, 0, stream>>>(out5, pooled, 128, 2, 1, 0);

  // 7) classifier head
  k_fc1<<<32, 128, 0, stream>>>(pooled, dnnwh, dnnb, fc1);
  k_fc2<<<4, 128, 0, stream>>>(fc1, outwh, outbp, logits);
  k_lsm<<<1, 256, 0, stream>>>(logits, (float*)d_out);
}
